// MultiHeadAttention_7129645711410
// MI455X (gfx1250) — matrix operand order repaired and checked by reading, not compiled, not run on hardware
//
#include <hip/hip_runtime.h>

#define NB 2
#define SS 4096
#define DD 512
#define HH 8
#define DHH 64
#define MM (NB * SS)  // 8192 rows

typedef __bf16 bf16_t;
typedef bf16_t v16bf __attribute__((ext_vector_type(16)));
typedef float v8f __attribute__((ext_vector_type(8)));
typedef unsigned int v4u __attribute__((ext_vector_type(4)));

union FragB16 {
  v16bf v;
  v4u q[2];
  unsigned short us[16];
};

__device__ __forceinline__ unsigned short f2bf(float f) {
  unsigned int u = __float_as_uint(f);
  u += 0x7FFFu + ((u >> 16) & 1u);  // round-to-nearest-even
  return (unsigned short)(u >> 16);
}

__device__ __forceinline__ float fast_exp2(float x) {
  return __builtin_amdgcn_exp2f(x);  // v_exp_f32 (native base-2)
}

__device__ __forceinline__ v8f wmma_bf16(v16bf a, v16bf b, v8f c) {
  return __builtin_amdgcn_wmma_f32_16x16x32_bf16(false, a, false, b, (short)0, c,
                                                 false, false);
}

// A-fragment (16x32 bf16): lane = M row (lane&15), g = lane>>4 selects K-halves.
__device__ __forceinline__ v16bf load_a_frag(const unsigned short* base, int g) {
  FragB16 f;
  f.q[0] = *(const v4u*)(base + g * 8);
  f.q[1] = *(const v4u*)(base + 16 + g * 8);
  return f.v;
}

// B-fragment (32x16 bf16) from a row-major [K x N] slab: p = &slab[k0][n], n = lane&15;
// element s holds k = 8g+s (s<8) or 16+8g+(s-8) (s>=8), row stride ld.
__device__ __forceinline__ v16bf load_b_frag_rowmajor(const unsigned short* p, int ld, int g) {
  FragB16 f;
#pragma unroll
  for (int s = 0; s < 8; ++s) {
    f.us[s] = p[(size_t)(8 * g + s) * ld];
    f.us[8 + s] = p[(size_t)(16 + 8 * g + s) * ld];
  }
  return f.v;
}

__device__ __forceinline__ float red_max16(float v) {
  v = fmaxf(v, __shfl_xor(v, 1, 32));
  v = fmaxf(v, __shfl_xor(v, 2, 32));
  v = fmaxf(v, __shfl_xor(v, 4, 32));
  v = fmaxf(v, __shfl_xor(v, 8, 32));
  return v;
}
__device__ __forceinline__ float red_sum16(float v) {
  v += __shfl_xor(v, 1, 32);
  v += __shfl_xor(v, 2, 32);
  v += __shfl_xor(v, 4, 32);
  v += __shfl_xor(v, 8, 32);
  return v;
}

// low 32 bits of a generic pointer to LDS == byte offset within wave LDS base
__device__ __forceinline__ unsigned lds_off(const void* p) {
  return (unsigned)(unsigned long long)p;
}

// per-lane async 16B copy global -> LDS (ASYNCcnt tracked)
__device__ __forceinline__ void async_ld_b128(unsigned ldsaddr, const void* g) {
  asm volatile("global_load_async_to_lds_b128 %0, %1, off" ::"v"(ldsaddr),
               "v"((unsigned long long)g)
               : "memory");
}
__device__ __forceinline__ void wait_async0() {
  asm volatile("s_wait_asynccnt 0x0" ::: "memory");
}

// ---------------------------------------------------------------- fp32 -> bf16
__global__ void k_convert(const float* __restrict__ src,
                          unsigned short* __restrict__ dst, int n4) {
  for (int i = blockIdx.x * blockDim.x + threadIdx.x; i < n4;
       i += gridDim.x * blockDim.x) {
    float4 f = ((const float4*)src)[i];
    unsigned long long p = (unsigned long long)f2bf(f.x) |
                           ((unsigned long long)f2bf(f.y) << 16) |
                           ((unsigned long long)f2bf(f.z) << 32) |
                           ((unsigned long long)f2bf(f.w) << 48);
    ((unsigned long long*)dst)[i] = p;
  }
}

// ------------------------------------------------- fused QKV projection + ReLU
// Each wave computes a 32x64 strip: B-frags shared across 2 row-tiles,
// A-frags shared across 4 col-tiles (0.5 b128 A loads + 8 u16 B loads per WMMA).
// K projection is written PRE-TRANSPOSED as [B,H,DH,S]; Q/V as [B,H,S,DH].
__global__ __launch_bounds__(256) void k_qkv(
    const unsigned short* __restrict__ xb, const unsigned short* __restrict__ wbase,
    const float* __restrict__ bq, const float* __restrict__ bk,
    const float* __restrict__ bv, unsigned short* __restrict__ qh,
    unsigned short* __restrict__ kt, unsigned short* __restrict__ vh) {
  const int lane = threadIdx.x & 31;
  const int wid = blockIdx.x * 8 + (threadIdx.x >> 5);
  const int per_proj = (MM / 32) * (DD / 64);  // 2048
  const int proj = wid / per_proj;
  const int t = wid % per_proj;
  const int m0 = (t / (DD / 64)) * 32;
  const int n0 = (t % (DD / 64)) * 64;

  const unsigned short* w = wbase + (size_t)proj * DD * DD;
  const float* bias = (proj == 0) ? bq : ((proj == 1) ? bk : bv);
  unsigned short* out = (proj == 0) ? qh : ((proj == 1) ? kt : vh);

  const int g = lane >> 4;
  const unsigned short* arow0 = xb + (size_t)(m0 + (lane & 15)) * DD;
  const unsigned short* arow1 = arow0 + (size_t)16 * DD;

  v8f acc[2][4];
#pragma unroll
  for (int mi = 0; mi < 2; ++mi)
#pragma unroll
    for (int ns = 0; ns < 4; ++ns) acc[mi][ns] = (v8f){};

#pragma unroll 2
  for (int k0 = 0; k0 < DD; k0 += 32) {
    v16bf a0 = load_a_frag(arow0 + k0, g);
    v16bf a1 = load_a_frag(arow1 + k0, g);
    const unsigned short* wrow = w + (size_t)k0 * DD + n0 + (lane & 15);
#pragma unroll
    for (int ns = 0; ns < 4; ++ns) {
      v16bf b = load_b_frag_rowmajor(wrow + ns * 16, DD, g);
      acc[0][ns] = wmma_bf16(a0, b, acc[0][ns]);
      acc[1][ns] = wmma_bf16(a1, b, acc[1][ns]);
    }
  }

  const int col = lane & 15;
#pragma unroll
  for (int ns = 0; ns < 4; ++ns) {
    const int n = n0 + ns * 16 + col;
    const int h = n >> 6;
    const int dh = n & 63;
    const float bv_ = bias[n];
#pragma unroll
    for (int mi = 0; mi < 2; ++mi) {
#pragma unroll
      for (int r = 0; r < 8; ++r) {
        const int m = m0 + mi * 16 + r + 8 * g;
        const int bi = m >> 12;
        const int s = m & 4095;
        const unsigned short v = f2bf(fmaxf(acc[mi][ns][r] + bv_, 0.0f));
        if (proj == 1)  // K stored d-major for transpose-free score fragments
          out[(((size_t)(bi * HH + h)) * DHH + dh) * SS + s] = v;
        else
          out[(((size_t)(bi * HH + h)) * SS + s) * DHH + dh] = v;
      }
    }
  }
}

// ------------------------------------------------------------- flash attention
// grid = B*H * (S/128); 8 waves/block, each wave owns 16 query rows.
// kt is [B*H, DH, S] (d-major), vh is [B*H, S, DH].
// Double-buffered async LDS staging overlaps the copy of tile j+1 with the
// compute of tile j; online softmax runs in exp2 domain.
__global__ __launch_bounds__(256) void k_attn(const unsigned short* __restrict__ qh,
                                              const unsigned short* __restrict__ kt,
                                              const unsigned short* __restrict__ vh,
                                              unsigned short* __restrict__ attn) {
  __shared__ __align__(16) unsigned short Kt[2][64 * 64];  // [d][key]
  __shared__ __align__(16) unsigned short Vt[2][64 * 64];  // [key][d]
  __shared__ __align__(16) unsigned short Ps[8][16 * 64];

  const int bh = blockIdx.x >> 5;  // b*H + h
  const int qtile = blockIdx.x & 31;
  const int wave = threadIdx.x >> 5;
  const int lane = threadIdx.x & 31;
  const int col = lane & 15;
  const int g = lane >> 4;
  const int q0 = qtile * 128 + wave * 16;

  // staging assignments for async tile copies (16 bf16 per thread per array)
  const int krow = threadIdx.x >> 2;          // d row 0..63
  const int kchunk = (threadIdx.x & 3) * 16;  // key offset within tile
  const unsigned short* ksrc = kt + ((size_t)bh * DHH + krow) * SS + kchunk;
  const unsigned short* vsrc = vh + (size_t)bh * SS * DHH + threadIdx.x * 16;
  const unsigned ka[2] = {lds_off(&Kt[0][krow * 64 + kchunk]),
                          lds_off(&Kt[1][krow * 64 + kchunk])};
  const unsigned va[2] = {lds_off(&Vt[0][threadIdx.x * 16]),
                          lds_off(&Vt[1][threadIdx.x * 16])};

  // Q fragments for this wave's 16 rows (DH=64 -> two 16x32 A-frags)
  const unsigned short* qrow = qh + ((size_t)bh * SS + q0 + col) * DHH;
  v16bf qf0 = load_a_frag(qrow, g);
  v16bf qf1 = load_a_frag(qrow + 32, g);

  v8f of[4];
  float mrow[8], lrow[8];
#pragma unroll
  for (int i = 0; i < 4; ++i) of[i] = (v8f){};
#pragma unroll
  for (int r = 0; r < 8; ++r) { mrow[r] = -3.0e38f; lrow[r] = 0.0f; }

  const float scale2 = 0.125f * 1.44269504088896340736f;  // log2(e)/sqrt(DH)

  // prologue: stage tile 0 into buffer 0
  async_ld_b128(ka[0], ksrc);
  async_ld_b128(ka[0] + 16, ksrc + 8);
  async_ld_b128(va[0], vsrc);
  async_ld_b128(va[0] + 16, vsrc + 8);
  wait_async0();
  __syncthreads();

  int cur = 0;
  for (int j0 = 0; j0 < SS; j0 += 64) {
    const int nxt = cur ^ 1;
    if (j0 + 64 < SS) {  // overlap: stage next tile while computing this one
      const int jn = j0 + 64;
      async_ld_b128(ka[nxt], ksrc + jn);
      async_ld_b128(ka[nxt] + 16, ksrc + jn + 8);
      async_ld_b128(va[nxt], vsrc + (size_t)jn * DHH);
      async_ld_b128(va[nxt] + 16, vsrc + (size_t)jn * DHH + 8);
      if (jn + 64 < SS) {  // L2 prefetch two tiles ahead
        __builtin_prefetch(ksrc + jn + 64, 0, 0);
        __builtin_prefetch(vsrc + (size_t)(jn + 64) * DHH, 0, 0);
      }
    }
    const unsigned short* Kc = Kt[cur];
    const unsigned short* Vc = Vt[cur];

    // scores: S = Q @ K^T; B-frag rows are contiguous in d-major Kc
    v8f sc[4];
#pragma unroll
    for (int jj = 0; jj < 4; ++jj) {
      sc[jj] = (v8f){};
#pragma unroll
      for (int c = 0; c < 2; ++c) {
        FragB16 bf;
        const unsigned short* krow_l = &Kc[(c * 32) * 64 + jj * 16 + col];
        bf.v = load_b_frag_rowmajor(krow_l, 64, g);
        sc[jj] = wmma_bf16(c ? qf1 : qf0, bf.v, sc[jj]);
      }
    }

    // online softmax (exp2 domain): rows r+8*g live in this 16-lane half
    float mnew[8], corr[8], rsum[8];
#pragma unroll
    for (int r = 0; r < 8; ++r) {
      float v = fmaxf(fmaxf(sc[0][r], sc[1][r]), fmaxf(sc[2][r], sc[3][r]));
      v = red_max16(v) * scale2;
      mnew[r] = fmaxf(mrow[r], v);
      corr[r] = fast_exp2(mrow[r] - mnew[r]);
      rsum[r] = 0.0f;
    }
#pragma unroll
    for (int jj = 0; jj < 4; ++jj) {
#pragma unroll
      for (int r = 0; r < 8; ++r) {
        float p = fast_exp2(sc[jj][r] * scale2 - mnew[r]);
        rsum[r] += p;
        Ps[wave][(r + 8 * g) * 64 + jj * 16 + col] = f2bf(p);
      }
    }
#pragma unroll
    for (int r = 0; r < 8; ++r) {
      lrow[r] = lrow[r] * corr[r] + red_sum16(rsum[r]);
      mrow[r] = mnew[r];
    }
#pragma unroll
    for (int ns = 0; ns < 4; ++ns)
#pragma unroll
      for (int r = 0; r < 8; ++r) of[ns][r] *= corr[r];

    // per-wave P scratch RAW: drain our ds stores before re-reading
    asm volatile("s_wait_dscnt 0x0" ::: "memory");

    // O += P @ V  (k-dim = 64 keys -> two 16x32 A-frags of P)
#pragma unroll
    for (int kk = 0; kk < 2; ++kk) {
      v16bf pf = load_a_frag(&Ps[wave][col * 64 + kk * 32], g);
#pragma unroll
      for (int ns = 0; ns < 4; ++ns) {
        v16bf vf = load_b_frag_rowmajor(&Vc[(kk * 32) * 64 + ns * 16 + col], 64, g);
        of[ns] = wmma_bf16(pf, vf, of[ns]);
      }
    }

    // next buffer staged + everyone done with this buffer
    wait_async0();
    __syncthreads();
    cur = nxt;
  }

  // normalize and write merged-head bf16 output [B,S,D]
  const int bi = bh >> 3;
  const int h = bh & 7;
#pragma unroll
  for (int r = 0; r < 8; ++r) {
    const float inv = 1.0f / lrow[r];
    const int srow = q0 + r + 8 * g;
#pragma unroll
    for (int ns = 0; ns < 4; ++ns)
      attn[((size_t)bi * SS + srow) * DD + h * DHH + ns * 16 + col] =
          f2bf(of[ns][r] * inv);
  }
}

// ----------------------------------------------- output projection + ReLU, f32
__global__ __launch_bounds__(256) void k_out(const unsigned short* __restrict__ a,
                                             const unsigned short* __restrict__ w,
                                             const float* __restrict__ bias,
                                             float* __restrict__ out) {
  const int lane = threadIdx.x & 31;
  const int wid = blockIdx.x * 8 + (threadIdx.x >> 5);
  const int m0 = (wid / (DD / 64)) * 32;
  const int n0 = (wid % (DD / 64)) * 64;
  const int g = lane >> 4;
  const unsigned short* arow0 = a + (size_t)(m0 + (lane & 15)) * DD;
  const unsigned short* arow1 = arow0 + (size_t)16 * DD;

  v8f acc[2][4];
#pragma unroll
  for (int mi = 0; mi < 2; ++mi)
#pragma unroll
    for (int ns = 0; ns < 4; ++ns) acc[mi][ns] = (v8f){};

#pragma unroll 2
  for (int k0 = 0; k0 < DD; k0 += 32) {
    v16bf a0 = load_a_frag(arow0 + k0, g);
    v16bf a1 = load_a_frag(arow1 + k0, g);
    const unsigned short* wrow = w + (size_t)k0 * DD + n0 + (lane & 15);
#pragma unroll
    for (int ns = 0; ns < 4; ++ns) {
      v16bf b = load_b_frag_rowmajor(wrow + ns * 16, DD, g);
      acc[0][ns] = wmma_bf16(a0, b, acc[0][ns]);
      acc[1][ns] = wmma_bf16(a1, b, acc[1][ns]);
    }
  }

  const int col = lane & 15;
#pragma unroll
  for (int ns = 0; ns < 4; ++ns) {
    const int n = n0 + ns * 16 + col;
    const float bv = bias[n];
#pragma unroll
    for (int mi = 0; mi < 2; ++mi)
#pragma unroll
      for (int r = 0; r < 8; ++r) {
        const int m = m0 + mi * 16 + r + 8 * g;
        out[(size_t)m * DD + n] = fmaxf(acc[mi][ns][r] + bv, 0.0f);
      }
  }
}

extern "C" void kernel_launch(void* const* d_in, const int* in_sizes, int n_in,
                              void* d_out, int out_size, void* d_ws,
                              size_t ws_size, hipStream_t stream) {
  const float* x = (const float*)d_in[0];
  const float* Wq = (const float*)d_in[1];
  const float* bq = (const float*)d_in[2];
  const float* Wk = (const float*)d_in[3];
  const float* bk = (const float*)d_in[4];
  const float* Wv = (const float*)d_in[5];
  const float* bv = (const float*)d_in[6];
  const float* Wo = (const float*)d_in[7];
  const float* bo = (const float*)d_in[8];

  unsigned short* ws = (unsigned short*)d_ws;
  unsigned short* xb = ws;                    // MM*DD
  unsigned short* wq = xb + (size_t)MM * DD;  // DD*DD each, x4
  unsigned short* wk = wq + (size_t)DD * DD;
  unsigned short* wv = wk + (size_t)DD * DD;
  unsigned short* wo = wv + (size_t)DD * DD;
  unsigned short* qh = wo + (size_t)DD * DD;  // MM*DD each, x3
  unsigned short* kt = qh + (size_t)MM * DD;  // stored [B,H,DH,S]
  unsigned short* vh = kt + (size_t)MM * DD;
  unsigned short* at = vh + (size_t)MM * DD;  // MM*DD

  k_convert<<<1024, 256, 0, stream>>>(x, xb, MM * DD / 4);
  k_convert<<<256, 256, 0, stream>>>(Wq, wq, DD * DD / 4);
  k_convert<<<256, 256, 0, stream>>>(Wk, wk, DD * DD / 4);
  k_convert<<<256, 256, 0, stream>>>(Wv, wv, DD * DD / 4);
  k_convert<<<256, 256, 0, stream>>>(Wo, wo, DD * DD / 4);

  // 3 * (8192/32) * (512/64) = 6144 waves -> 768 blocks of 8 waves
  k_qkv<<<768, 256, 0, stream>>>(xb, wq, bq, bk, bv, qh, kt, vh);
  // B*H * (S/128) = 512 blocks
  k_attn<<<512, 256, 0, stream>>>(qh, kt, vh, at);
  // 2048 waves -> 256 blocks
  k_out<<<256, 256, 0, stream>>>(at, wo, bo, (float*)d_out);
}
